// SCRC_v2_78254304133873
// MI455X (gfx1250) — matrix operand order repaired and checked by reading, not compiled, not run on hardware
//
#include <hip/hip_runtime.h>
#include <hip/hip_bf16.h>

#define BATCH  4096
#define DIMK   128
#define UNITS  32768
#define KSEL   32

typedef __attribute__((ext_vector_type(16))) __bf16          v16bf;
typedef __attribute__((ext_vector_type(8)))  float           v8f;
typedef __attribute__((ext_vector_type(16))) unsigned short  v16u;
typedef __attribute__((ext_vector_type(8)))  unsigned short  v8u;
typedef __attribute__((ext_vector_type(4)))  unsigned short  v4u;
typedef __attribute__((ext_vector_type(4)))  float           f4;

static __device__ __forceinline__ unsigned short f2bf(float f) {
  unsigned u = __builtin_bit_cast(unsigned, f);
  u += 0x7FFFu + ((u >> 16) & 1u);          // round-to-nearest-even
  return (unsigned short)(u >> 16);
}

// ---------------- x row L2-normalize: fp32 out + bf16 workspace copy -------
__global__ __launch_bounds__(128) void xnorm_kernel(const float* __restrict__ x,
                                                    float* __restrict__ xn,
                                                    unsigned short* __restrict__ xb) {
  const int row = blockIdx.x;
  const int t   = threadIdx.x;               // 128 threads = 4 waves
  float v = x[row * DIMK + t];
  float s = v * v;
  #pragma unroll
  for (int off = 16; off; off >>= 1) s += __shfl_xor(s, off, 32);
  __shared__ float ps[4];
  if ((t & 31) == 0) ps[t >> 5] = s;
  __syncthreads();
  const float tot = ps[0] + ps[1] + ps[2] + ps[3];
  const float inv = 1.0f / fmaxf(sqrtf(tot), 1e-12f);
  const float xv  = v * inv;
  xn[row * DIMK + t] = xv;
  xb[row * DIMK + t] = f2bf(xv);
}

// ---------------- W fp32 -> bf16 workspace ---------------------------------
__global__ __launch_bounds__(256) void wconv_kernel(const float* __restrict__ W,
                                                    unsigned short* __restrict__ Wb) {
  const int i = (blockIdx.x * 256 + threadIdx.x) * 4;
  f4 w = *(const f4*)(W + i);
  v4u o;
  o.x = f2bf(w.x); o.y = f2bf(w.y); o.z = f2bf(w.z); o.w = f2bf(w.w);
  *(v4u*)(Wb + i) = o;
}

// ---------------- zero-fill z with wide non-temporal stores ----------------
__global__ __launch_bounds__(256) void zfill_kernel(f4* __restrict__ z) {
  const size_t i = ((size_t)blockIdx.x * 256 + threadIdx.x) * 2;
  f4 zero = {0.f, 0.f, 0.f, 0.f};
  __builtin_nontemporal_store(zero, z + i);
  __builtin_nontemporal_store(zero, z + i + 1);
}

// ---------------- fused bf16-WMMA GEMM + per-row partial top-32 ------------
// grid: (128 row-blocks of 32 rows, 4 column groups of 8192 units), 256 thr.
// Each step covers 128 units: every wave computes TWO 16x16 tiles; B for the
// next step is issued right after the WMMAs so its latency hides under the
// LDS staging + barriers + selection phase.
__global__ __launch_bounds__(256) void scores_topk_kernel(
    const unsigned short* __restrict__ Wb,
    const unsigned short* __restrict__ xb,
    float* __restrict__ pv, int* __restrict__ pi) {
  const int lane   = threadIdx.x & 31;
  const int w      = threadIdx.x >> 5;       // wave 0..7
  const int rt     = w >> 2;                 // row tile 0..1
  const int ct     = w & 3;                  // col tile 0..3
  const int rowblk = blockIdx.x;             // 0..127
  const int colg   = blockIdx.y;             // 0..3

  // ---- A (x_norm bf16), 16x32 layout: lanes<16 K{0..15},
  //      lanes>=16 K{16..31}; 4 chunks cover K=128, kept in VGPRs ----
  const int arow = rowblk * 32 + rt * 16 + (lane & 15);
  const int koff = (lane & 16) ? 16 : 0;
  v16u a[4];
  #pragma unroll
  for (int c = 0; c < 4; ++c) {
    const unsigned short* p = xb + arow * DIMK + c * 32 + koff;
    v8u lo = *(const v8u*)(p);
    v8u hi = *(const v8u*)(p + 8);
    a[c] = __builtin_shufflevector(lo, hi, 0,1,2,3,4,5,6,7,8,9,10,11,12,13,14,15);
  }

  __shared__ float sc[32][128];

  float rv[4], rmin4[4];
  int   ri[4];
  #pragma unroll
  for (int r = 0; r < 4; ++r) { rv[r] = -3.0e38f; ri[r] = 0; rmin4[r] = -3.0e38f; }

  const int bk = ((lane >> 4) & 1) * 16;     // B: lanes<16 K0..15, lanes>=16 K16..31
  const int bn = lane & 15;

  // B base pointer for this wave's first sub-tile (cols ct*16 .. ct*16+15);
  // second sub-tile is +64 units; next step is +128 units.
  const unsigned short* bp =
      Wb + (size_t)(colg * 8192 + ct * 16 + bn) * DIMK + bk;

  v16u b0[4], b1[4];
  #pragma unroll
  for (int c = 0; c < 4; ++c) {
    b0[c] = *(const v16u*)(bp + c * 32);
    b1[c] = *(const v16u*)(bp + (size_t)64 * DIMK + c * 32);
  }

  for (int step = 0; step < 8192 / 128; ++step) {
    const int cbase = colg * 8192 + step * 128;       // first unit of this slab

    v8f acc0 = {0.f, 0.f, 0.f, 0.f, 0.f, 0.f, 0.f, 0.f};
    v8f acc1 = {0.f, 0.f, 0.f, 0.f, 0.f, 0.f, 0.f, 0.f};
    #pragma unroll
    for (int c = 0; c < 4; ++c)
      acc0 = __builtin_amdgcn_wmma_f32_16x16x32_bf16(
                 false, __builtin_bit_cast(v16bf, a[c]),
                 false, __builtin_bit_cast(v16bf, b0[c]),
                 (short)0, acc0, false, false);
    #pragma unroll
    for (int c = 0; c < 4; ++c)
      acc1 = __builtin_amdgcn_wmma_f32_16x16x32_bf16(
                 false, __builtin_bit_cast(v16bf, a[c]),
                 false, __builtin_bit_cast(v16bf, b1[c]),
                 (short)0, acc1, false, false);

    // issue next step's B now: latency hides under LDS + barriers + selection
    if (step + 1 < 8192 / 128) {
      const unsigned short* np = bp + (size_t)(step + 1) * 128 * DIMK;
      __builtin_prefetch(np + (size_t)128 * DIMK, 0, 3);   // step+2, WGP scope
      #pragma unroll
      for (int c = 0; c < 4; ++c) {
        b0[c] = *(const v16u*)(np + c * 32);
        b1[c] = *(const v16u*)(np + (size_t)64 * DIMK + c * 32);
      }
    }

    // C layout: lanes 0-15 -> M=i, lanes 16-31 -> M=i+8; N = lane&15
    {
      const int mrow = rt * 16 + ((lane >> 4) << 3);
      const int col0 = ct * 16 + bn;
      #pragma unroll
      for (int i = 0; i < 8; ++i) sc[mrow + i][col0] = acc0[i];
      #pragma unroll
      for (int i = 0; i < 8; ++i) sc[mrow + i][col0 + 64] = acc1[i];
    }
    __syncthreads();

    // wave w owns rows w*4 .. w*4+3; ballot-filter + wave argmin replacement.
    // candidate col = j*32 + lane for j = 0..3 (conflict-free strided reads)
    #pragma unroll
    for (int r = 0; r < 4; ++r) {
      const int lrow = w * 4 + r;
      const float m  = rmin4[r];
      const float v0 = sc[lrow][lane];
      const float v1 = sc[lrow][lane + 32];
      const float v2 = sc[lrow][lane + 64];
      const float v3 = sc[lrow][lane + 96];
      unsigned long long mks[2];
      mks[0] = (unsigned long long)__ballot(v0 > m) |
               ((unsigned long long)__ballot(v1 > m) << 32);
      mks[1] = (unsigned long long)__ballot(v2 > m) |
               ((unsigned long long)__ballot(v3 > m) << 32);
      #pragma unroll
      for (int h = 0; h < 2; ++h) {
        unsigned long long mk = mks[h];
        const int colbase = h * 64;
        while (mk) {
          const int b = __ffsll((unsigned long long)mk) - 1;
          mk &= mk - 1;
          const int col = colbase + b;               // bit b <-> col b in half
          const float v = sc[lrow][col];             // uniform broadcast
          float mv = rv[r]; int ml = lane;           // wave argmin of list
          #pragma unroll
          for (int off = 16; off; off >>= 1) {
            const float ov = __shfl_xor(mv, off, 32);
            const int   ol = __shfl_xor(ml, off, 32);
            if (ov < mv || (ov == mv && ol < ml)) { mv = ov; ml = ol; }
          }
          if (v > mv) {
            if (lane == ml) { rv[r] = v; ri[r] = cbase + col; }
            float nm = rv[r];
            #pragma unroll
            for (int off = 16; off; off >>= 1)
              nm = fminf(nm, __shfl_xor(nm, off, 32));
            rmin4[r] = nm;
          } else {
            rmin4[r] = mv;
          }
        }
      }
    }
    __syncthreads();
  }

  // emit partial top-32 list (one entry per lane)
  #pragma unroll
  for (int r = 0; r < 4; ++r) {
    const int grow = rowblk * 32 + w * 4 + r;
    const size_t o = ((size_t)grow * 4 + colg) * 32 + lane;
    pv[o] = rv[r];
    pi[o] = ri[r];
  }
}

// ---------------- merge 4 partials/row, sort, scatter into z ---------------
__global__ __launch_bounds__(256) void topk_merge_kernel(
    const float* __restrict__ pv, const int* __restrict__ pi,
    float* __restrict__ z, float* __restrict__ idx_out) {
  const int lane = threadIdx.x & 31;
  const int w    = threadIdx.x >> 5;
  const int row  = blockIdx.x * 8 + w;       // one wave per row

  __shared__ float cv[8][128];
  __shared__ int   ci[8][128];
  #pragma unroll
  for (int g = 0; g < 4; ++g) {
    const size_t o = ((size_t)row * 4 + g) * 32 + lane;
    cv[w][g * 32 + lane] = pv[o];
    ci[w][g * 32 + lane] = pi[o];
  }
  // same-wave LDS producer/consumer: DScnt keeps order, no cross-wave share.
  float lv = cv[w][lane];
  int   li = ci[w][lane];
  float mn = lv;
  #pragma unroll
  for (int off = 16; off; off >>= 1) mn = fminf(mn, __shfl_xor(mn, off, 32));

  for (int b = 32; b < 128; ++b) {
    const float v = cv[w][b];                // uniform
    if (v <= mn) continue;                   // uniform branch
    const int vi = ci[w][b];
    float mv = lv; int ml = lane;
    #pragma unroll
    for (int off = 16; off; off >>= 1) {
      const float ov = __shfl_xor(mv, off, 32);
      const int   ol = __shfl_xor(ml, off, 32);
      if (ov < mv || (ov == mv && ol < ml)) { mv = ov; ml = ol; }
    }
    if (lane == ml) { lv = v; li = vi; }
    mn = lv;
    #pragma unroll
    for (int off = 16; off; off >>= 1) mn = fminf(mn, __shfl_xor(mn, off, 32));
  }

  // bitonic sort 32 lanes, descending, tie-break on lower index
  #pragma unroll
  for (int k = 2; k <= 32; k <<= 1) {
    #pragma unroll
    for (int j = k >> 1; j > 0; j >>= 1) {
      const float ov = __shfl_xor(lv, j, 32);
      const int   oi = __shfl_xor(li, j, 32);
      const bool up        = (lane & k) != 0;
      const bool lower     = (lane & j) == 0;
      const bool takeLarge = (lower != up);
      const bool ogreater  = (ov > lv) || (ov == lv && oi < li);
      if (ogreater == takeLarge) { lv = ov; li = oi; }
    }
  }

  z[(size_t)row * UNITS + li] = fmaxf(lv, 0.0f);
  idx_out[row * KSEL + lane]  = (float)li;
}

extern "C" void kernel_launch(void* const* d_in, const int* in_sizes, int n_in,
                              void* d_out, int out_size, void* d_ws, size_t ws_size,
                              hipStream_t stream) {
  const float* x = (const float*)d_in[0];
  const float* W = (const float*)d_in[1];
  // d_in[2] = k (constant 32)

  float* z   = (float*)d_out;
  float* xn  = z  + (size_t)BATCH * UNITS;
  float* idx = xn + (size_t)BATCH * DIMK;

  char* ws = (char*)d_ws;
  unsigned short* Wb = (unsigned short*)ws;                                  // 8 MB
  unsigned short* xb = (unsigned short*)(ws + (size_t)UNITS * DIMK * 2);     // 1 MB
  float* pv = (float*)(ws + (size_t)UNITS * DIMK * 2 + (size_t)BATCH * DIMK * 2); // 2 MB
  int*   pi = (int*)((char*)pv + (size_t)BATCH * 4 * 32 * 4);                // 2 MB

  wconv_kernel<<<(UNITS * DIMK) / (256 * 4), 256, 0, stream>>>(W, Wb);
  xnorm_kernel<<<BATCH, DIMK, 0, stream>>>(x, xn, xb);
  zfill_kernel<<<((size_t)BATCH * UNITS / 4) / (256 * 2), 256, 0, stream>>>((f4*)z);
  scores_topk_kernel<<<dim3(128, 4), 256, 0, stream>>>(Wb, xb, pv, pi);
  topk_merge_kernel<<<BATCH / 8, 256, 0, stream>>>(pv, pi, z, idx);
}
